// CrossTransformerBlock_21603685499516
// MI455X (gfx1250) — compile-verified, matrix operand order checked by reading, not run on hardware
//
#include <hip/hip_runtime.h>

#define Bn   4
#define NQn  4096
#define Nn   2048
#define DIMn 128
#define DGn  256
#define DIn  256
#define Kn   16

typedef __attribute__((ext_vector_type(16))) _Float16 v16h;
typedef __attribute__((ext_vector_type(8)))  float    v8f;

// ---- CDNA5 WMMA 16x16x32 f16 fragment layout helpers (ISA 7.12.2) ----
// lane L: half-group h = L>>4, row/col = L&15.
// element pair p (0..7) maps to K-base:
__device__ __host__ __forceinline__ int kb_of(int p, int h) {
  return (p < 4) ? (8 * h + 2 * p) : (16 + 8 * h + 2 * (p - 4));
}

// A-matrix fragment (M x K tile, A row-major [32][128] f16 in LDS)
__device__ __forceinline__ v16h load_a_frag(const _Float16* A, int mtile, int kstep, int lane) {
  const int h = (lane >> 4) & 1;
  const int m = mtile * 16 + (lane & 15);
  const _Float16* row = A + m * DIMn + kstep * 32;
  v16h a;
#pragma unroll
  for (int p = 0; p < 8; ++p) {
    const int kb = kb_of(p, h);
    a[2 * p]     = row[kb];
    a[2 * p + 1] = row[kb + 1];
  }
  return a;
}

// ---------------- tiny per-batch global q/k/v projections ----------------
__global__ void globals_kernel(const float* __restrict__ lat_rep,
                               const float* __restrict__ W_q,
                               const float* __restrict__ W_kg,
                               const float* __restrict__ W_vg,
                               float* __restrict__ qkv /* [3][B*128] */) {
  const int t = blockIdx.x * blockDim.x + threadIdx.x;
  if (t >= Bn * DIMn) return;
  const int o = t & (DIMn - 1);
  const int b = t >> 7;
  const float* lr = lat_rep + b * DGn;
  float aq = 0.f, ak = 0.f, av = 0.f;
  for (int c = 0; c < DGn; ++c) {
    const float x = lr[c];
    aq += x * W_q [o * DGn + c];
    ak += x * W_kg[o * DGn + c];
    av += x * W_vg[o * DGn + c];
  }
  qkv[0 * (Bn * DIMn) + t] = aq;
  qkv[1 * (Bn * DIMn) + t] = ak;
  qkv[2 * (Bn * DIMn) + t] = av;
}

// -------- pre-swizzle 128x128 weights into per-lane B-matrix fragments --------
// dst layout: [w(3)][nt(8)][ks(4)][lane(32)][16 halves] ; B[k][n] = W[n][k]
__global__ void swizzle_wb_kernel(const float* __restrict__ Wd2,
                                  const float* __restrict__ Wg1,
                                  const float* __restrict__ Wg2,
                                  _Float16* __restrict__ dst) {
  const int t = blockIdx.x * blockDim.x + threadIdx.x;
  if (t >= 3 * 1024) return;
  const int wsel = t >> 10;
  const int r    = t & 1023;         // (nt*4+ks)*32 + lane
  const int lane = r & 31;
  const int ks   = (r >> 5) & 3;
  const int nt   = r >> 7;
  const float* W = (wsel == 0) ? Wd2 : (wsel == 1) ? Wg1 : Wg2;
  _Float16* o = dst + wsel * 16384 + r * 16;
  const int h = (lane >> 4) & 1;
  const int n = nt * 16 + (lane & 15);
#pragma unroll
  for (int p = 0; p < 8; ++p) {
    const int k = ks * 32 + kb_of(p, h);
    o[2 * p]     = (_Float16)W[n * DIMn + k];
    o[2 * p + 1] = (_Float16)W[n * DIMn + k + 1];
  }
}

// ---------------- k_all / v_all projections (points @ W^T), f16 out ----------------
__global__ void proj_kernel(const float* __restrict__ points,
                            const float* __restrict__ W_k,
                            const float* __restrict__ W_v,
                            _Float16* __restrict__ k_all,
                            _Float16* __restrict__ v_all) {
  const int t = blockIdx.x * blockDim.x + threadIdx.x;
  if (t >= Bn * Nn * DIMn) return;
  const int o  = t & (DIMn - 1);
  const int bn = t >> 7;
  const float* pt = points + (size_t)bn * DIn;
  float ak = 0.f, av = 0.f;
  for (int c = 0; c < DIn; ++c) {
    const float x = pt[c];
    ak += x * W_k[o * DIn + c];
    av += x * W_v[o * DIn + c];
  }
  k_all[(size_t)bn * DIMn + o] = (_Float16)ak;
  v_all[(size_t)bn * DIMn + o] = (_Float16)av;
}

// ---------------- kNN: one wave32 per query ----------------
__global__ __launch_bounds__(256) void knn_kernel(const float* __restrict__ xyz_q,
                                                  const float* __restrict__ xyz,
                                                  int* __restrict__ knn_idx) {
  __shared__ float sxyz[Nn * 3];                // 24 KB
  __shared__ float sd[8][32][Kn];               // 16 KB
  __shared__ int   si[8][32][Kn];               // 16 KB

  const int q0g  = blockIdx.x * 8;              // first global query of block
  const int b    = q0g >> 12;                   // /NQ
  const int q0   = q0g & (NQn - 1);
  const int lane = threadIdx.x & 31;
  const int wave = threadIdx.x >> 5;

  for (int i = threadIdx.x; i < Nn * 3; i += 256)
    sxyz[i] = xyz[(size_t)b * Nn * 3 + i];
  __syncthreads();

  const int q  = q0 + wave;
  const int gq = b * NQn + q;
  const float qx = xyz_q[gq * 3 + 0];
  const float qy = xyz_q[gq * 3 + 1];
  const float qz = xyz_q[gq * 3 + 2];

  float bd[Kn]; int bi[Kn];
#pragma unroll
  for (int j = 0; j < Kn; ++j) { bd[j] = 3.4e38f; bi[j] = 0; }

  for (int p = lane; p < Nn; p += 32) {
    const float dx = qx - sxyz[p * 3 + 0];
    const float dy = qy - sxyz[p * 3 + 1];
    const float dz = qz - sxyz[p * 3 + 2];
    float dd = dx * dx + dy * dy + dz * dz;
    int   ii = p;
    if (dd < bd[Kn - 1]) {
#pragma unroll
      for (int j = 0; j < Kn; ++j) {           // sorted ascending insert
        if (dd < bd[j]) {
          const float td = bd[j]; const int ti = bi[j];
          bd[j] = dd; bi[j] = ii; dd = td; ii = ti;
        }
      }
    }
  }
#pragma unroll
  for (int j = 0; j < Kn; ++j) { sd[wave][lane][j] = bd[j]; si[wave][lane][j] = bi[j]; }
  __syncthreads();

  // 16-round wave merge of 32 sorted lists
  int ptr = 0;
  for (int r = 0; r < Kn; ++r) {
    const float head = (ptr < Kn) ? sd[wave][lane][ptr] : 3.4e38f;
    const int   hidx = si[wave][lane][(ptr < Kn) ? ptr : 0];
    float m = head;
#pragma unroll
    for (int off = 16; off > 0; off >>= 1) m = fminf(m, __shfl_xor(m, off, 32));
    const unsigned long long msk = __ballot(head == m);
    const int winner = __ffsll(msk) - 1;
    const int widx   = __shfl(hidx, winner, 32);
    if (lane == winner) ptr++;
    if (lane == 0) knn_idx[(size_t)gq * Kn + r] = widx;
  }
}

// ---------------- fused attention: one 4-wave workgroup per query ----------------
__global__ __launch_bounds__(128) void attn_wmma_kernel(
    const float* __restrict__ xyz_q, const float* __restrict__ xyz,
    const int* __restrict__ knn, const _Float16* __restrict__ k_all,
    const _Float16* __restrict__ v_all, const float* __restrict__ qkv,
    const float* __restrict__ W_d1, const float* __restrict__ b_d1,
    const float* __restrict__ b_d2, const float* __restrict__ b_g1,
    const float* __restrict__ b_g2,
    const _Float16* __restrict__ Wd2_sw, const _Float16* __restrict__ Wg1_sw,
    const _Float16* __restrict__ Wg2_sw,
    float* __restrict__ out) {
  __shared__ _Float16 sA[32 * DIMn];     // layer A operand (f16)
  __shared__ _Float16 sH[32 * DIMn];     // hidden activations (f16)
  __shared__ float    sPos[Kn * DIMn];   // pos encoding (f32)
  __shared__ float    sAttn[17 * DIMn];  // pre-softmax scores (f32)
  __shared__ int      sIdx[Kn];
  __shared__ float    sRel[Kn * 3];

  const int gq   = blockIdx.x;
  const int b    = gq >> 12;
  const int t    = threadIdx.x;          // 0..127, one column per thread
  const int lane = t & 31;
  const int wave = t >> 5;

  if (t < Kn) sIdx[t] = knn[(size_t)gq * Kn + t];
  __syncthreads();
  if (t < Kn * 3) {
    const int r = t / 3, c = t - 3 * r;
    sRel[t] = xyz_q[gq * 3 + c] - xyz[((size_t)b * Nn + sIdx[r]) * 3 + c];
  }
  __syncthreads();

  // ---- fc_delta layer1 (3->128) + ReLU, VALU ----
  {
    const int c = t;
    const float w0 = W_d1[c * 3 + 0], w1 = W_d1[c * 3 + 1], w2 = W_d1[c * 3 + 2];
    const float bb = b_d1[c];
#pragma unroll
    for (int r = 0; r < Kn; ++r) {
      const float v = fmaxf(w0 * sRel[r * 3] + w1 * sRel[r * 3 + 1] + w2 * sRel[r * 3 + 2] + bb, 0.f);
      sA[r * DIMn + c] = (_Float16)v;
    }
#pragma unroll
    for (int r = Kn; r < 32; ++r) sA[r * DIMn + c] = (_Float16)0.f;
  }
  __syncthreads();

  // ---- fc_delta layer2 (128->128) via WMMA, M-tile 0 only: wave -> ntiles {2w,2w+1} ----
  {
#pragma unroll
    for (int i = 0; i < 2; ++i) {
      const int nt = wave * 2 + i;
      const int nn = nt * 16 + (lane & 15);
      const float bb = b_d2[nn];
      v8f acc;
#pragma unroll
      for (int v = 0; v < 8; ++v) acc[v] = bb;
#pragma unroll
      for (int ks = 0; ks < 4; ++ks) {
        const v16h a  = load_a_frag(sA, 0, ks, lane);
        const v16h bf = ((const v16h*)Wd2_sw)[(nt * 4 + ks) * 32 + lane];
        acc = __builtin_amdgcn_wmma_f32_16x16x32_f16(false, a, false, bf, (short)0, acc, false, false);
      }
      const int h = (lane >> 4) & 1;
#pragma unroll
      for (int v = 0; v < 8; ++v) sPos[(v + 8 * h) * DIMn + nn] = acc[v];
    }
  }
  __syncthreads();

  // ---- build s = q - k + pos rows (17 valid, pad to 32) ----
  {
    const int c = t;
    const float qv = qkv[0 * (Bn * DIMn) + b * DIMn + c];
    const float kg = qkv[1 * (Bn * DIMn) + b * DIMn + c];
#pragma unroll
    for (int r = 0; r < Kn; ++r) {
      const float kv = (float)k_all[((size_t)b * Nn + sIdx[r]) * DIMn + c];
      sA[r * DIMn + c] = (_Float16)(qv - kv + sPos[r * DIMn + c]);
    }
    sA[Kn * DIMn + c] = (_Float16)(qv - kg);       // global row, pos = 0
#pragma unroll
    for (int r = Kn + 1; r < 32; ++r) sA[r * DIMn + c] = (_Float16)0.f;
  }
  __syncthreads();

  // ---- fc_gamma layer1 (128->128) + ReLU via WMMA, M=32 ----
  // wave mapping: mtile = wave&1, ntiles = (wave>>1)*4 .. +3
  {
    const int mt = wave & 1;
#pragma unroll
    for (int i = 0; i < 4; ++i) {
      const int nt = (wave >> 1) * 4 + i;
      const int nn = nt * 16 + (lane & 15);
      const float bb = b_g1[nn];
      v8f acc;
#pragma unroll
      for (int v = 0; v < 8; ++v) acc[v] = bb;
#pragma unroll
      for (int ks = 0; ks < 4; ++ks) {
        const v16h a  = load_a_frag(sA, mt, ks, lane);
        const v16h bf = ((const v16h*)Wg1_sw)[(nt * 4 + ks) * 32 + lane];
        acc = __builtin_amdgcn_wmma_f32_16x16x32_f16(false, a, false, bf, (short)0, acc, false, false);
      }
      const int h = (lane >> 4) & 1;
#pragma unroll
      for (int v = 0; v < 8; ++v) {
        const int m = mt * 16 + v + 8 * h;
        sH[m * DIMn + nn] = (_Float16)fmaxf(acc[v], 0.f);
      }
    }
  }
  __syncthreads();

  // ---- fc_gamma layer2 (128->128) via WMMA, keep rows 0..16 ----
  {
    const int mt = wave & 1;
#pragma unroll
    for (int i = 0; i < 4; ++i) {
      const int nt = (wave >> 1) * 4 + i;
      const int nn = nt * 16 + (lane & 15);
      const float bb = b_g2[nn];
      v8f acc;
#pragma unroll
      for (int v = 0; v < 8; ++v) acc[v] = bb;
#pragma unroll
      for (int ks = 0; ks < 4; ++ks) {
        const v16h a  = load_a_frag(sH, mt, ks, lane);
        const v16h bf = ((const v16h*)Wg2_sw)[(nt * 4 + ks) * 32 + lane];
        acc = __builtin_amdgcn_wmma_f32_16x16x32_f16(false, a, false, bf, (short)0, acc, false, false);
      }
      const int h = (lane >> 4) & 1;
#pragma unroll
      for (int v = 0; v < 8; ++v) {
        const int m = mt * 16 + v + 8 * h;
        if (m < Kn + 1) sAttn[m * DIMn + nn] = acc[v];
      }
    }
  }
  __syncthreads();

  // ---- per-column softmax over 17 rows + weighted sum of (v + pos) ----
  {
    const int c = t;
    float vals[Kn + 1];
    float mx = -3.4e38f;
#pragma unroll
    for (int r = 0; r < Kn + 1; ++r) { vals[r] = sAttn[r * DIMn + c]; mx = fmaxf(mx, vals[r]); }
    float sum = 0.f;
#pragma unroll
    for (int r = 0; r < Kn + 1; ++r) { vals[r] = expf(vals[r] - mx); sum += vals[r]; }
    const float inv = 1.f / sum;
    const float vg  = qkv[2 * (Bn * DIMn) + b * DIMn + c];
    float accv = vals[Kn] * inv * vg;                 // global row (pos = 0)
#pragma unroll
    for (int r = 0; r < Kn; ++r) {
      const float vv = (float)v_all[((size_t)b * Nn + sIdx[r]) * DIMn + c];
      accv += vals[r] * inv * (vv + sPos[r * DIMn + c]);
    }
    out[(size_t)gq * DIMn + c] = accv;
  }
}

// ---------------- launcher ----------------
extern "C" void kernel_launch(void* const* d_in, const int* in_sizes, int n_in,
                              void* d_out, int out_size, void* d_ws, size_t ws_size,
                              hipStream_t stream) {
  (void)in_sizes; (void)n_in; (void)out_size; (void)ws_size;
  const float* xyz_q  = (const float*)d_in[0];
  const float* lat    = (const float*)d_in[1];
  const float* xyz    = (const float*)d_in[2];
  const float* points = (const float*)d_in[3];
  const float* W_d1   = (const float*)d_in[4];
  const float* b_d1   = (const float*)d_in[5];
  const float* W_d2   = (const float*)d_in[6];
  const float* b_d2   = (const float*)d_in[7];
  const float* W_g1   = (const float*)d_in[8];
  const float* b_g1   = (const float*)d_in[9];
  const float* W_g2   = (const float*)d_in[10];
  const float* b_g2   = (const float*)d_in[11];
  const float* W_kg   = (const float*)d_in[12];
  const float* W_vg   = (const float*)d_in[13];
  const float* W_q    = (const float*)d_in[14];
  const float* W_k    = (const float*)d_in[15];
  const float* W_v    = (const float*)d_in[16];
  float* out = (float*)d_out;

  char* ws = (char*)d_ws;
  int*      knn   = (int*)ws;                                   // 1 MiB
  _Float16* k_all = (_Float16*)(ws + (4u << 20));               // 2 MiB
  _Float16* v_all = (_Float16*)(ws + (6u << 20));               // 2 MiB
  float*    qkv   = (float*)(ws + (8u << 20));                  // 6 KiB
  _Float16* wsw   = (_Float16*)(ws + (8u << 20) + 8192);        // 96 KiB

  globals_kernel<<<2, 256, 0, stream>>>(lat, W_q, W_kg, W_vg, qkv);
  swizzle_wb_kernel<<<12, 256, 0, stream>>>(W_d2, W_g1, W_g2, wsw);
  proj_kernel<<<(Bn * Nn * DIMn + 255) / 256, 256, 0, stream>>>(points, W_k, W_v, k_all, v_all);
  knn_kernel<<<(Bn * NQn) / 8, 256, 0, stream>>>(xyz_q, xyz, knn);
  attn_wmma_kernel<<<Bn * NQn, 128, 0, stream>>>(
      xyz_q, xyz, knn, k_all, v_all, qkv,
      W_d1, b_d1, b_d2, b_g1, b_g2,
      wsw, wsw + 16384, wsw + 32768, out);
}
